// GRU_Dynamic_49624052138329
// MI455X (gfx1250) — matrix operand order repaired and checked by reading, not compiled, not run on hardware
//
#include <hip/hip_runtime.h>
#include <cstdint>
#include <cstddef>

// ---- problem dims ----
#define BDIM   64
#define TDIM   512
#define HDIM   512
#define NLAT   256
#define NCOND  64
#define IN0    320           // NLAT + NCOND
#define G3H    1536          // 3*HDIM
#define BT     (BDIM*TDIM)   // 32768
#define NWG    8             // workgroups in the sequential step kernel
#define LDPAD  40            // LDS tile row pitch in halves (80B: 16B-aligned, conflict-free)

typedef __attribute__((ext_vector_type(16))) __bf16 v16bf;
typedef __attribute__((ext_vector_type(8)))  float  v8f;

// ---------------------------------------------------------------------------
// CDNA5 async copy: global -> LDS, 16 bytes per lane, tracked by ASYNCcnt.
// ---------------------------------------------------------------------------
__device__ __forceinline__ void async_cp16(uint32_t lds_off, const __bf16* g) {
  asm volatile("global_load_async_to_lds_b128 %0, %1, off"
               : : "v"(lds_off), "v"(g) : "memory");
}
__device__ __forceinline__ void wait_async0() {
  asm volatile("s_wait_asynccnt 0x0" ::: "memory");
}
__device__ __forceinline__ uint32_t lds_addr(const void* p) {
  return (uint32_t)(uintptr_t)p;   // low 32 bits of generic addr == LDS offset
}

// ---------------------------------------------------------------------------
// WMMA fragment loaders (wave32, V_WMMA_F32_16X16X32_BF16; k-contiguous order used for both A and B)
//
// A (16x32, MxK):
//   lane<16 : row=lane,    K[0..15]
//   lane>=16: row=lane-16, K[16..31]
// B (32x16, KxN), stored as W[N,K] row-major (K contiguous):
//   lane<16 : col=lane,    K[0..15];  lane>=16: col=lane-16, K[16..31]
// ---------------------------------------------------------------------------
__device__ __forceinline__ v16bf load_a_frag(const __bf16* X, int ldx,
                                             int row0, int k0, int lane) {
  int m  = lane & 15;
  int ko = (lane >> 4) << 4;           // 0 or 16
  const __bf16* p = X + (size_t)(row0 + m) * ldx + (k0 + ko);
  union { uint4 u[2]; v16bf v; } r;
  r.u[0] = *reinterpret_cast<const uint4*>(p);
  r.u[1] = *reinterpret_cast<const uint4*>(p + 8);
  return r.v;
}
__device__ __forceinline__ v16bf load_b_frag(const __bf16* W, int ldk,
                                             int n0, int k0, int lane) {
  int n  = lane & 15;
  int ko = (lane >> 4) << 4;           // 0 or 16
  const __bf16* p = W + (size_t)(n0 + n) * ldk + (k0 + ko);
  union { uint4 u[2]; v16bf v; } r;
  r.u[0] = *reinterpret_cast<const uint4*>(p);
  r.u[1] = *reinterpret_cast<const uint4*>(p + 8);
  return r.v;
}

// same layouts but from an LDS tile with row pitch LDPAD halves
__device__ __forceinline__ v16bf a_frag_lds(const __bf16* T, int row0, int lane) {
  int m  = row0 + (lane & 15);
  int ko = (lane >> 4) << 4;
  const __bf16* p = T + m * LDPAD + ko;
  union { uint4 u[2]; v16bf v; } r;
  r.u[0] = *reinterpret_cast<const uint4*>(p);
  r.u[1] = *reinterpret_cast<const uint4*>(p + 8);
  return r.v;
}
__device__ __forceinline__ v16bf b_frag_lds(const __bf16* T, int row0, int lane) {
  int n  = row0 + (lane & 15);
  int ko = (lane >> 4) << 4;
  const __bf16* p = T + n * LDPAD + ko;
  union { uint4 u[2]; v16bf v; } r;
  r.u[0] = *reinterpret_cast<const uint4*>(p);
  r.u[1] = *reinterpret_cast<const uint4*>(p + 8);
  return r.v;
}

// ---------------------------------------------------------------------------
// GEMM: out[M,N](f32) = X[M,K](bf16, ld=K) * W^T (W is [N,K] bf16) + bias.
// Block: 256 thr = 8 waves; block tile 128(M) x 128(N); wave tile 32 x 64
// (2 M-subtiles x 4 N-subtiles = 8 WMMA per K-step; B frag reused twice).
// K staged 32 at a time into double-buffered LDS via async copies.
// Grid: (M/128, N/128).  K must be a multiple of 32.
// ---------------------------------------------------------------------------
__global__ void __launch_bounds__(256)
gemm_bias_kernel(const __bf16* __restrict__ X, const __bf16* __restrict__ W,
                 const float* __restrict__ bias, float* __restrict__ out,
                 int K, int N) {
  __shared__ __align__(16) __bf16 Asm[2][128 * LDPAD];
  __shared__ __align__(16) __bf16 Bsm[2][128 * LDPAD];

  const int tid  = threadIdx.x;
  const int lane = tid & 31;
  const int w    = tid >> 5;
  const int mblk = blockIdx.x * 128;
  const int nblk = blockIdx.y * 128;

  // staging: A and B tiles are each 128 rows x 4 chunks(8h) = 512 x 16B
  auto stage = [&](int buf, int k0) {
#pragma unroll
    for (int s = 0; s < 2; ++s) {
      int chunk = tid + s * 256;
      int row = chunk >> 2, ck = (chunk & 3) * 8;
      async_cp16(lds_addr(&Asm[buf][row * LDPAD + ck]),
                 X + (size_t)(mblk + row) * K + k0 + ck);
      async_cp16(lds_addr(&Bsm[buf][row * LDPAD + ck]),
                 W + (size_t)(nblk + row) * K + k0 + ck);
    }
  };

  const int mt = (w & 3) * 32;                       // wave 32-row tile
  const int ng = (w >> 2) * 64;                      // wave 64-col group

  stage(0, 0);
  wait_async0();
  __syncthreads();

  v8f acc[8] = {};                                   // [n-subtile][m-subtile]
  const int KT = K >> 5;
  for (int kt = 0; kt < KT; ++kt) {
    const int cur = kt & 1;
    if (kt + 1 < KT) stage(cur ^ 1, (kt + 1) * 32);  // overlap with WMMAs

    v16bf a0 = a_frag_lds(Asm[cur], mt, lane);
    v16bf a1 = a_frag_lds(Asm[cur], mt + 16, lane);
#pragma unroll
    for (int j = 0; j < 4; ++j) {
      v16bf b = b_frag_lds(Bsm[cur], ng + 16 * j, lane);
      acc[2 * j]     = __builtin_amdgcn_wmma_f32_16x16x32_bf16(
          false, a0, false, b, (short)0, acc[2 * j], false, false);
      acc[2 * j + 1] = __builtin_amdgcn_wmma_f32_16x16x32_bf16(
          false, a1, false, b, (short)0, acc[2 * j + 1], false, false);
    }
    wait_async0();
    __syncthreads();
  }

  // C/D layout: VGPR r -> M = r + (lane>=16 ? 8 : 0), N = lane&15
  const int rbase = mblk + mt + ((lane >> 4) << 3);
  const int cbase = lane & 15;
#pragma unroll
  for (int j = 0; j < 4; ++j) {
    int col = nblk + ng + 16 * j + cbase;
    float bv = bias[col];
#pragma unroll
    for (int s = 0; s < 2; ++s)
#pragma unroll
      for (int r = 0; r < 8; ++r)
        out[(size_t)(rbase + 16 * s + r) * N + col] = acc[2 * j + s][r] + bv;
  }
}

// ---------------------------------------------------------------------------
// Device-scope split barrier over the NWG persistent workgroups.
// ---------------------------------------------------------------------------
__device__ __forceinline__ void grid_bar(unsigned int* ctr, unsigned int target) {
  __threadfence();
  __syncthreads();
  if (threadIdx.x == 0) {
    __hip_atomic_fetch_add(ctr, 1u, __ATOMIC_RELEASE, __HIP_MEMORY_SCOPE_AGENT);
    while (__hip_atomic_load(ctr, __ATOMIC_ACQUIRE, __HIP_MEMORY_SCOPE_AGENT) < target)
      __builtin_amdgcn_s_sleep(1);
  }
  __syncthreads();
  __threadfence();
}

// ---------------------------------------------------------------------------
// Sequential GRU layer: 8 persistent WGs, each owns 64 h-columns and computes
// the r/z/n gate columns for its slice each step via WMMA, then the pointwise
// GRU update. h kept f32 in LDS (owner), bf16 ping-pong in global (for A-frags).
// gi = x@W_ih^T + b_ih precomputed ([BT, 1536], row = b*T + t).
// ---------------------------------------------------------------------------
__global__ void __launch_bounds__(256)
gru_step_kernel(const __bf16* __restrict__ Whh,    // [1536,512] bf16 (orig layout)
                const float*  __restrict__ b_hh,   // [1536]
                const float*  __restrict__ gi,     // [BT,1536] f32 (has b_ih)
                const float*  __restrict__ h0,     // [64,512] f32
                __bf16*       __restrict__ hbuf,   // [2][64*512] bf16 ping-pong
                __bf16*       __restrict__ x_out,  // [BT,512] bf16 layer output
                float*        __restrict__ h_last, // [64,512] f32 (d_out slice)
                unsigned int* __restrict__ ctr) {
  __shared__ float hsl[64 * 64];                   // f32 h slice owned by this WG

  const int tid  = threadIdx.x;
  const int lane = tid & 31;
  const int w    = tid >> 5;
  const int c0   = blockIdx.x * 64;                // owned h-column base

  // seed h state (f32 LDS slice + bf16 global buf0)
  for (int i = tid; i < 64 * 64; i += 256) {
    int row = i >> 6, col = i & 63;
    float v = h0[row * HDIM + c0 + col];
    hsl[i] = v;
    hbuf[row * HDIM + c0 + col] = (__bf16)v;
  }
  grid_bar(ctr, NWG * 1u);

  const int mt = w & 3;                            // wave row tile (16 rows)
  const int g  = w >> 2;                           // wave 32-col subgroup
  const int s0 = c0 + g * 32;

  for (int t = 0; t < TDIM; ++t) {
    const __bf16* hcur = hbuf + (size_t)(t & 1) * (64 * HDIM);
    __bf16*       hnxt = hbuf + (size_t)((t + 1) & 1) * (64 * HDIM);

    v8f acc[6] = {};                               // {r,z,n} x {2 col-halves}
    for (int k0 = 0; k0 < HDIM; k0 += 32) {
      v16bf a = load_a_frag(hcur, HDIM, mt * 16, k0, lane);
#pragma unroll
      for (int q = 0; q < 6; ++q) {
        int gate = q >> 1, half = q & 1;
        v16bf b = load_b_frag(Whh, HDIM, gate * HDIM + s0 + 16 * half, k0, lane);
        acc[q] = __builtin_amdgcn_wmma_f32_16x16x32_bf16(
            false, a, false, b, (short)0, acc[q], false, false);
      }
    }

    const int rbase = mt * 16 + ((lane >> 4) << 3);
#pragma unroll
    for (int half = 0; half < 2; ++half) {
      int clocal = g * 32 + 16 * half + (lane & 15);   // 0..63 within slice
      int colg   = c0 + clocal;                        // global h column
      float bhr = b_hh[colg];
      float bhz = b_hh[HDIM + colg];
      float bhn = b_hh[2 * HDIM + colg];
#pragma unroll
      for (int r = 0; r < 8; ++r) {
        int row = rbase + r;                           // batch index
        size_t gir = ((size_t)row * TDIM + t) * (size_t)G3H;
        float ir  = gi[gir + colg];
        float iz  = gi[gir + HDIM + colg];
        float in_ = gi[gir + 2 * HDIM + colg];
        float ghr = acc[0 + half][r] + bhr;
        float ghz = acc[2 + half][r] + bhz;
        float ghn = acc[4 + half][r] + bhn;
        float rg = 1.0f / (1.0f + __expf(-(ir + ghr)));
        float ug = 1.0f / (1.0f + __expf(-(iz + ghz)));
        float ng = tanhf(in_ + rg * ghn);
        float hp = hsl[row * 64 + clocal];
        float hn = (1.0f - ug) * ng + ug * hp;
        hsl[row * 64 + clocal] = hn;
        hnxt[row * HDIM + colg] = (__bf16)hn;
        x_out[((size_t)row * TDIM + t) * HDIM + colg] = (__bf16)hn;
      }
    }
    grid_bar(ctr, NWG * (unsigned)(t + 2));
  }

  __syncthreads();
  for (int i = tid; i < 64 * 64; i += 256)
    h_last[(i >> 6) * HDIM + c0 + (i & 63)] = hsl[i];
}

// ---------------------------------------------------------------------------
// helpers: f32 -> bf16 copies, input concat, counter reset
// ---------------------------------------------------------------------------
__global__ void f2bf_kernel(const float* __restrict__ src,
                            __bf16* __restrict__ dst, int n) {
  int i = blockIdx.x * 256 + threadIdx.x;
  if (i < n) dst[i] = (__bf16)src[i];
}

__global__ void xcat_kernel(const float* __restrict__ z,
                            const float* __restrict__ a,
                            __bf16* __restrict__ dst) {
  int i = blockIdx.x * 256 + threadIdx.x;
  if (i >= BT * IN0) return;
  int row = i / IN0, c = i - row * IN0;
  float v = (c < NLAT) ? z[(size_t)row * NLAT + c]
                       : a[(size_t)row * NCOND + (c - NLAT)];
  dst[i] = (__bf16)v;
}

__global__ void zero_ctr_kernel(unsigned int* c) {
  if (threadIdx.x < 8) c[threadIdx.x] = 0u;
}

// ---------------------------------------------------------------------------
extern "C" void kernel_launch(void* const* d_in, const int* in_sizes, int n_in,
                              void* d_out, int out_size, void* d_ws, size_t ws_size,
                              hipStream_t stream) {
  (void)in_sizes; (void)n_in; (void)out_size; (void)ws_size;

  const float* z   = (const float*)d_in[0];
  const float* a   = (const float*)d_in[1];
  const float* h0  = (const float*)d_in[2];                 // [L,B,H]
  const float* Wih[3] = {(const float*)d_in[3],  (const float*)d_in[7],  (const float*)d_in[11]};
  const float* Whh[3] = {(const float*)d_in[4],  (const float*)d_in[8],  (const float*)d_in[12]};
  const float* bih[3] = {(const float*)d_in[5],  (const float*)d_in[9],  (const float*)d_in[13]};
  const float* bhh[3] = {(const float*)d_in[6],  (const float*)d_in[10], (const float*)d_in[14]};
  const float* fcw = (const float*)d_in[15];
  const float* fcb = (const float*)d_in[16];

  float* outz = (float*)d_out;                               // [B,T,256]
  float* outh = (float*)d_out + (size_t)BT * NLAT;           // [L,B,H]

  // workspace carve (256B aligned)
  char* ws = (char*)d_ws;
  size_t off = 0;
  auto carve = [&](size_t bytes) -> char* {
    char* p = ws + off;
    off = (off + bytes + 255) & ~(size_t)255;
    return p;
  };
  __bf16* wih_b[3]; __bf16* whh_b[3];
  wih_b[0] = (__bf16*)carve((size_t)G3H * IN0  * 2);
  whh_b[0] = (__bf16*)carve((size_t)G3H * HDIM * 2);
  wih_b[1] = (__bf16*)carve((size_t)G3H * HDIM * 2);
  whh_b[1] = (__bf16*)carve((size_t)G3H * HDIM * 2);
  wih_b[2] = (__bf16*)carve((size_t)G3H * HDIM * 2);
  whh_b[2] = (__bf16*)carve((size_t)G3H * HDIM * 2);
  __bf16* wfc_b = (__bf16*)carve((size_t)NLAT * HDIM * 2);
  __bf16* xcat  = (__bf16*)carve((size_t)BT * IN0  * 2);
  __bf16* xA    = (__bf16*)carve((size_t)BT * HDIM * 2);
  __bf16* xB    = (__bf16*)carve((size_t)BT * HDIM * 2);
  float*  gi    = (float*) carve((size_t)BT * G3H * 4);
  __bf16* hbuf  = (__bf16*)carve((size_t)2 * BDIM * HDIM * 2);
  unsigned int* ctr = (unsigned int*)carve(64);

  // 1) weights -> bf16 (layouts preserved: W is [N,K] row-major == B-frag layout)
  auto conv = [&](const float* s, __bf16* d, int n) {
    f2bf_kernel<<<dim3((n + 255) / 256), dim3(256), 0, stream>>>(s, d, n);
  };
  conv(Wih[0], wih_b[0], G3H * IN0);
  conv(Whh[0], whh_b[0], G3H * HDIM);
  conv(Wih[1], wih_b[1], G3H * HDIM);
  conv(Whh[1], whh_b[1], G3H * HDIM);
  conv(Wih[2], wih_b[2], G3H * HDIM);
  conv(Whh[2], whh_b[2], G3H * HDIM);
  conv(fcw,    wfc_b,    NLAT * HDIM);

  // 2) concat(z,a) -> bf16
  xcat_kernel<<<dim3((BT * IN0 + 255) / 256), dim3(256), 0, stream>>>(z, a, xcat);

  // 3) reset device barrier counters
  zero_ctr_kernel<<<dim3(1), dim3(32), 0, stream>>>(ctr);

  // 4) per layer: wide gi GEMM, then persistent sequential step kernel
  const __bf16* xin[3]  = {xcat, xA, xB};
  __bf16*       xout[3] = {xA, xB, xA};
  const int     kdim[3] = {IN0, HDIM, HDIM};
  for (int l = 0; l < 3; ++l) {
    gemm_bias_kernel<<<dim3(BT / 128, G3H / 128), dim3(256), 0, stream>>>(
        xin[l], wih_b[l], bih[l], gi, kdim[l], G3H);
    gru_step_kernel<<<dim3(NWG), dim3(256), 0, stream>>>(
        whh_b[l], bhh[l], gi, h0 + (size_t)l * BDIM * HDIM,
        hbuf, xout[l], outh + (size_t)l * BDIM * HDIM, ctr + l);
  }

  // 5) final FC: z_ = x2 @ fc_w^T + fc_b
  gemm_bias_kernel<<<dim3(BT / 128, NLAT / 128), dim3(256), 0, stream>>>(
      xA, wfc_b, fcb, outz, HDIM, NLAT);
}
